// CRF_42271068127559
// MI455X (gfx1250) — matrix operand order repaired and checked by reading, not compiled, not run on hardware
//
#include <hip/hip_runtime.h>
#include <hip/hip_bf16.h>

// CRF mean-field kernel for MI455X (gfx1250, wave32).
// One workgroup (256 threads = 8 waves) per batch element.
//  * feats staged to LDS via GLOBAL_LOAD_ASYNC_TO_LDS_B128 (ASYNCcnt path)
//  * Gram matrix via v_wmma_f32_16x16x32_bf16 with hi/lo bf16 error-split
//  * only upper-triangular tiles computed (G is symmetric), mirrored in LDS

typedef __attribute__((ext_vector_type(16))) __bf16 v16bf;
typedef __attribute__((ext_vector_type(4)))  __bf16 bf4;
typedef __attribute__((ext_vector_type(8)))  float  v8f;

#define NN   64
#define EMB  128
#define ITER 10
#define LDP  65   // padded LDS row stride (bank-conflict-free matvec + mirror)

union FragU { v16bf v; uint4 q[2]; };

// A-operand (16x32 bf16): lanes 0-15 -> K {kc+0..7, kc+16..23},
//                         lanes16-31 -> K {kc+8..15, kc+24..31}
__device__ __forceinline__ v16bf load_frag_a(const __bf16* base, int row, int kc, int lane) {
    const int khalf = (lane >> 4) << 3;            // 0 or 8
    const __bf16* p = base + row * EMB + kc + khalf;
    FragU f;
    f.q[0] = *(const uint4*)(p);                   // 8 bf16: K .. K+7
    f.q[1] = *(const uint4*)(p + 16);              // 8 bf16: K+16 .. K+23
    return f.v;
}

// B-operand (32x16 bf16, column n = lane&15): lanes 0-15 -> K {kc+0..7, kc+16..23},
//                                             lanes16-31 -> K {kc+8..15, kc+24..31}
__device__ __forceinline__ v16bf load_frag_b(const __bf16* base, int row, int kc, int lane) {
    const int koff = (lane >> 4) << 3;             // 0 or 8
    const __bf16* p = base + row * EMB + kc + koff;
    FragU f;
    f.q[0] = *(const uint4*)(p);                   // K .. K+7
    f.q[1] = *(const uint4*)(p + 16);              // K+16 .. K+23
    return f.v;
}

__global__ __launch_bounds__(256, 2)
void crf_meanfield_kernel(const float* __restrict__ feats,
                          const float* __restrict__ logits,
                          const float* __restrict__ W,
                          float* __restrict__ out)
{
    __shared__ __align__(16) float  stageA[NN * EMB];  // 32 KB async landing pad
    __shared__ __align__(16) __bf16 hiA[NN * EMB];     // 16 KB
    __shared__ __align__(16) __bf16 loA[NN * EMB];     // 16 KB
    __shared__ __align__(16) float  Gm[NN * LDP];      // 16.25 KB (G, then PP)
    __shared__ __align__(16) float  Wsym[NN * LDP];    // 16.25 KB
    __shared__ float lvec[NN];
    __shared__ float tvec[NN];
    __shared__ float rn[NN];

    const int b    = blockIdx.x;
    const int tid  = threadIdx.x;
    const int lane = tid & 31;
    const int wave = tid >> 5;

    const float* fb = feats + (size_t)b * NN * EMB;

    // ---- Async global -> LDS staging of the 64x128 fp32 tile -------------
    // 8 x B128 per thread; no VGPR results, tracked on ASYNCcnt.
    {
        const unsigned lds0 = (unsigned)(size_t)(void*)&stageA[0];
#pragma unroll
        for (int q = 0; q < (NN * EMB / 4) / 256; ++q) {   // 8 chunks / thread
            const int v = tid + q * 256;
            const unsigned long long gaddr =
                (unsigned long long)(size_t)(((const float4*)fb) + v);
            const unsigned lofs = lds0 + (unsigned)v * 16u;
            asm volatile("global_load_async_to_lds_b128 %0, %1, off"
                         :: "v"(lofs), "v"(gaddr) : "memory");
        }
    }

    // ---- Overlap the async transfer: Wsym (L2-resident) + logits ---------
    for (int idx = tid; idx < NN * NN; idx += 256) {
        int n = idx >> 6, m = idx & 63;
        Wsym[n * LDP + m] = 0.5f * (W[idx] + W[m * NN + n]);
    }
    float u = 0.f;
    if (tid < NN) {
        u = logits[(size_t)b * NN + tid];
        lvec[tid] = u;
    }

    asm volatile("s_wait_asynccnt 0x0" ::: "memory");
    __syncthreads();

    // ---- Split fp32 -> (hi, lo) bf16 planes (LDS -> LDS) -----------------
    for (int v = tid; v < NN * EMB / 4; v += 256) {
        float4 f = ((const float4*)stageA)[v];
        bf4 h, l;
        h.x = (__bf16)f.x; l.x = (__bf16)(f.x - (float)h.x);
        h.y = (__bf16)f.y; l.y = (__bf16)(f.y - (float)h.y);
        h.z = (__bf16)f.z; l.z = (__bf16)(f.z - (float)h.z);
        h.w = (__bf16)f.w; l.w = (__bf16)(f.w - (float)h.w);
        ((bf4*)hiA)[v] = h;
        ((bf4*)loA)[v] = l;
    }
    __syncthreads();

    // ---- Gram matrix G = A * A^T via WMMA (upper-triangular tiles) -------
    // 10 tiles (i<=j) over 8 waves: wave w gets tile w; waves 0,1 also get 8,9.
    // Split product: A*A^T ~= H*H^T + H*L^T + L*H^T  (3 bf16 WMMAs / k-step)
    const int ntile = (wave < 2) ? 2 : 1;
    for (int q = 0; q < ntile; ++q) {
        const int t = wave + q * 8;
        // nibble-packed (i,j) for tiles 0..9: 00 01 02 03 11 12 13 22 23 33
        const unsigned enc = (t < 8)
            ? (unsigned)((0x2213121103020100ull >> (8 * t)) & 0xFF)
            : ((t == 8) ? 0x23u : 0x33u);
        const int ti = (int)(enc >> 4);
        const int tj = (int)(enc & 0xF);
        const int ra = ti * 16 + (lane & 15);
        const int rb = tj * 16 + (lane & 15);
        v8f acc = {0.f, 0.f, 0.f, 0.f, 0.f, 0.f, 0.f, 0.f};
#pragma unroll
        for (int kc = 0; kc < EMB; kc += 32) {
            v16bf ah = load_frag_a(hiA, ra, kc, lane);
            v16bf al = load_frag_a(loA, ra, kc, lane);
            v16bf bh = load_frag_b(hiA, rb, kc, lane);
            v16bf bl = load_frag_b(loA, rb, kc, lane);
            acc = __builtin_amdgcn_wmma_f32_16x16x32_bf16(
                      false, ah, false, bh, (short)0, acc, false, false);
            acc = __builtin_amdgcn_wmma_f32_16x16x32_bf16(
                      false, ah, false, bl, (short)0, acc, false, false);
            acc = __builtin_amdgcn_wmma_f32_16x16x32_bf16(
                      false, al, false, bh, (short)0, acc, false, false);
        }
        // C/D layout: VGPR v, lanes 0-15 -> M=v, N=lane; lanes 16-31 -> M=v+8
        const int n0 = ti * 16 + ((lane >> 4) << 3);
        const int m  = tj * 16 + (lane & 15);
#pragma unroll
        for (int v = 0; v < 8; ++v)
            Gm[(n0 + v) * LDP + m] = acc[v];
        if (ti != tj) {            // mirror (G symmetric); uniform branch
#pragma unroll
            for (int v = 0; v < 8; ++v)
                Gm[m * LDP + (n0 + v)] = acc[v];
        }
    }
    __syncthreads();

    // ---- Norms come free from the Gram diagonal --------------------------
    if (tid < NN) rn[tid] = rsqrtf(Gm[tid * LDP + tid]);
    __syncthreads();

    // ---- PP = (G / (|n||m|)) * Wsym  (in place) --------------------------
    for (int idx = tid; idx < NN * NN; idx += 256) {
        int n = idx >> 6, m = idx & 63;
        Gm[n * LDP + m] *= rn[n] * rn[m] * Wsym[n * LDP + m];
    }
    __syncthreads();

    // ---- 10 mean-field iterations: l = u + PP * tanh(l/2) ----------------
    for (int it = 0; it < ITER; ++it) {
        if (tid < NN) tvec[tid] = tanhf(0.5f * lvec[tid]);   // == 2*sigmoid(l)-1
        __syncthreads();
        if (tid < NN) {
            const float* row = &Gm[tid * LDP];
            float s = u;
#pragma unroll 8
            for (int m = 0; m < NN; ++m) s += row[m] * tvec[m];
            lvec[tid] = s;
        }
        __syncthreads();
    }

    if (tid < NN) out[(size_t)b * NN + tid] = lvec[tid];
}

extern "C" void kernel_launch(void* const* d_in, const int* in_sizes, int n_in,
                              void* d_out, int out_size, void* d_ws, size_t ws_size,
                              hipStream_t stream) {
    const float* feats  = (const float*)d_in[0];   // [B, 64, 128] fp32
    const float* logits = (const float*)d_in[1];   // [B, 64, 1]   fp32
    const float* W      = (const float*)d_in[2];   // [1, 64, 64]  fp32
    float* out          = (float*)d_out;           // [B, 64, 1]   fp32

    const int batch = in_sizes[1] / NN;            // 8192
    crf_meanfield_kernel<<<dim3(batch), dim3(256), 0, stream>>>(feats, logits, W, out);
}
